// Qwen2_5_VisionTransformer_58265526338304
// MI455X (gfx1250) — matrix operand order repaired and checked by reading, compiled, not run on hardware
//
#include <hip/hip_runtime.h>

// ---------------------------------------------------------------------------
// Qwen2.5 ViT forward for MI455X (gfx1250, wave32).
//   * all heavy GEMMs on v_wmma_f32_16x16x32_bf16 (f32 accumulate)
//   * A tiles staged global->LDS by the Tensor Data Mover (tensor_load_to_lds,
//     TENSORcnt / s_wait_tensorcnt), with hardware LDS padding matching the
//     padded fragment layout
//   * B tiles staged manually (transposed) + global_prefetch for L2 priming
// ---------------------------------------------------------------------------

typedef unsigned short u16;
typedef __attribute__((ext_vector_type(16))) __bf16 bf16x16;
typedef __attribute__((ext_vector_type(8)))  float   f32x8;
typedef __attribute__((ext_vector_type(4)))  unsigned int u32x4;
typedef __attribute__((ext_vector_type(8)))  int i32x8;
typedef __attribute__((ext_vector_type(4)))  int i32x4;

#if defined(__gfx1250__) && __has_builtin(__builtin_amdgcn_tensor_load_to_lds) && \
    __has_builtin(__builtin_amdgcn_s_wait_tensorcnt)
#define HAVE_TDM 1
#else
#define HAVE_TDM 0
#endif

#if __has_include(<hip/amd_detail/amd_gfx1250_TDM.h>)
#define TDM_6ARG 1
#else
#define TDM_6ARG 0
#endif

__device__ __forceinline__ u16 f2bf(float f) {
  unsigned int u = __float_as_uint(f);
  u += 0x7fffu + ((u >> 16) & 1u);        // round-to-nearest-even
  return (u16)(u >> 16);
}

// ---------------- zero fill (graph-capture-safe memset) ----------------
__global__ void fill_zero_kernel(float* p, size_t n) {
  size_t i = (size_t)blockIdx.x * blockDim.x + threadIdx.x;
  size_t stride = (size_t)gridDim.x * blockDim.x;
  for (; i < n; i += stride) p[i] = 0.f;
}

// ---------------- f32 -> bf16 convert ----------------
__global__ void cvt_bf16_kernel(const float* __restrict__ in, u16* __restrict__ out, size_t n) {
  size_t i = (size_t)blockIdx.x * blockDim.x + threadIdx.x;
  size_t stride = (size_t)gridDim.x * blockDim.x;
  for (; i < n; i += stride) out[i] = f2bf(in[i]);
}

// ---------------- RMSNorm (+ optional residual add) ----------------
__global__ __launch_bounds__(256)
void rmsnorm_kernel(const float* x, const float* resid, const float* w,
                    float* hOut, u16* nOut) {
  const int H = 1280;
  int row = blockIdx.x;
  int tid = threadIdx.x;
  __shared__ float red[256];
  float hv[5];
  float ss = 0.f;
#pragma unroll
  for (int i = 0; i < 5; ++i) {
    int c = tid + i * 256;
    size_t idx = (size_t)row * H + c;
    float v = x[idx];
    if (resid) v += resid[idx];
    hv[i] = v;
    ss += v * v;
  }
  red[tid] = ss;
  __syncthreads();
  for (int s = 128; s > 0; s >>= 1) {
    if (tid < s) red[tid] += red[tid + s];
    __syncthreads();
  }
  float scale = rsqrtf(red[0] * (1.f / H) + 1e-6f);
#pragma unroll
  for (int i = 0; i < 5; ++i) {
    int c = tid + i * 256;
    size_t idx = (size_t)row * H + c;
    if (hOut) hOut[idx] = hv[i];
    nOut[idx] = f2bf(hv[i] * scale * w[c]);
  }
}

// ---------------- windowed attention with fused RoPE (flash-style) ----------------
__global__ __launch_bounds__(64)
void attn_kernel(const float* qkv, const float* cosp, const float* sinp, u16* ob) {
  const int H = 1280, HD = 80, QKVW = 3840;
  int w = blockIdx.x >> 4;
  int h = blockIdx.x & 15;
  __shared__ float Qs[64][HD];
  __shared__ float Ks[64][HD];
  __shared__ float Vs[64][HD];
  int t = threadIdx.x;
  int tok = w * 64 + t;
  const float* base = qkv + (size_t)tok * QKVW + h * HD;
#pragma unroll 4
  for (int d = 0; d < HD; ++d) {
    float c = cosp[(size_t)tok * HD + d];
    float s = sinp[(size_t)tok * HD + d];
    float qd = base[d];
    float kd = base[1280 + d];
    float qr = (d < 40) ? -base[d + 40] : base[d - 40];
    float kr = (d < 40) ? -base[1280 + d + 40] : base[1280 + d - 40];
    Qs[t][d] = qd * c + qr * s;
    Ks[t][d] = kd * c + kr * s;
    Vs[t][d] = base[2560 + d];
  }
  __syncthreads();
  const float scale = 0.11180339887498949f;  // 1/sqrt(80)
  float m = -3.4e38f, l = 0.f;
  float o[HD];
#pragma unroll
  for (int d = 0; d < HD; ++d) o[d] = 0.f;
  for (int c = 0; c < 64; ++c) {
    float dot = 0.f;
#pragma unroll
    for (int d = 0; d < HD; ++d) dot += Qs[t][d] * Ks[c][d];
    float s = dot * scale;
    float mn = fmaxf(m, s);
    float corr = __expf(m - mn);
    float p = __expf(s - mn);
    l = l * corr + p;
#pragma unroll
    for (int d = 0; d < HD; ++d) o[d] = o[d] * corr + p * Vs[c][d];
    m = mn;
  }
  float inv = 1.f / l;
#pragma unroll
  for (int d = 0; d < HD; ++d)
    ob[(size_t)tok * H + h * HD + d] = f2bf(o[d] * inv);
}

// ---------------- bf16 WMMA GEMM, 128x128x32 tiles, fused epilogues ----------------
// mode 0: outF = acc + bias
// mode 1: outF = acc + bias + aux                (residual add)
// mode 2: outB = bf16( silu(aux) * (acc+bias) )  (SwiGLU: aux = gate f32)
// mode 3: outB = bf16( gelu(acc+bias) )          (exact gelu)
#define LDP 40  // padded LDS K-stride in halves (80 B rows; TDM pad 4 dw / 16 dw)

__global__ __launch_bounds__(256)
void gemm_bf16_wmma(const u16* __restrict__ A, const u16* __restrict__ B,
                    const float* __restrict__ bias, const float* aux,
                    float* outF, u16* outB, int M, int N, int K, int mode) {
  __shared__ __align__(16) u16 As[2][128 * LDP];
  __shared__ __align__(16) u16 Bs[2][128 * LDP];
  int tid = threadIdx.x;
  int wave = tid >> 5, lane = tid & 31;
  int lr = lane & 15, hi = lane >> 4;
  int wm = wave >> 2, wn = wave & 3;  // waves: 2 (M) x 4 (N)
  int mBase = blockIdx.y * 128, nBase = blockIdx.x * 128;

  f32x8 acc[4][2] = {};  // 64x32 per wave = 4x2 wmma tiles

#if HAVE_TDM
  // ---- A tile via Tensor Data Mover: 2D tile 32(K) x 128(M), bf16,
  //      LDS pad of 4 DWORDs after every 16 DWORDs -> row stride 80 B = LDP.
  auto loadTileA = [&](int s, int kb) {
    if (tid < 32) {  // wave 0 issues the DMA; TDM ignores EXEC, gate per-wave
      unsigned ldsAddr =
          (unsigned)(size_t)(__attribute__((address_space(3))) u16*)&As[s][0];
      unsigned long long ga = (unsigned long long)(size_t)(A + (size_t)mBase * K + kb);
      u32x4 g0;
      g0[0] = 1u;                                   // count=1, user desc
      g0[1] = ldsAddr;                              // lds_addr [63:32]
      g0[2] = (unsigned)(ga & 0xffffffffull);       // global_addr lo
      g0[3] = (unsigned)((ga >> 32) & 0x1ffffffull) // global_addr hi (57b)
              | (2u << 30);                         // type=2 (image)
      unsigned td0 = (unsigned)(K - kb);            // remaining K (zero-fill OOB)
      unsigned td1 = (unsigned)(M - mBase);         // remaining M
      unsigned long long st0 = (unsigned long long)K;
      i32x8 g1;
      g1[0] = (int)((1u << 16)        // data_size = 2 bytes
                    | (1u << 20)      // pad_enable
                    | (3u << 22)      // pad_interval: 16 DWORDs
                    | (3u << 25));    // pad_amount:   4 DWORDs
      g1[1] = (int)((td0 & 0xffffu) << 16);                    // tensor_dim0 lo16
      g1[2] = (int)((td0 >> 16) | ((td1 & 0xffffu) << 16));    // dim0 hi / dim1 lo
      g1[3] = (int)((td1 >> 16) | (32u << 16));                // dim1 hi, tile_dim0=32
      g1[4] = (int)128;                                        // tile_dim1=128
      g1[5] = (int)(st0 & 0xffffffffull);                      // dim0_stride lo
      g1[6] = (int)((st0 >> 32) & 0xffffull);                  // dim0_stride hi
      g1[7] = 0;
      i32x4 z4 = {0, 0, 0, 0};
#if TDM_6ARG
      i32x8 z8 = {0, 0, 0, 0, 0, 0, 0, 0};
      __builtin_amdgcn_tensor_load_to_lds(g0, g1, z4, z4, z8, 0);
#else
      __builtin_amdgcn_tensor_load_to_lds(g0, g1, z4, z4, 0);
#endif
    }
  };
#else
  auto loadTileA = [&](int s, int kb) {
    for (int c = tid; c < 512; c += 256) {
      int row = c >> 2, kc = (c & 3) << 3;
      int gm = mBase + row, gk = kb + kc;
      union { u16 h[8]; uint2 v2[2]; uint4 v4; } tmp;
      size_t eoff = (size_t)gm * K + gk;
      if (gm < M && gk + 8 <= K && ((eoff & 3) == 0)) {
        const u16* p = A + eoff;
        tmp.v2[0] = *(const uint2*)p;
        tmp.v2[1] = *(const uint2*)(p + 4);
      } else {
        for (int j = 0; j < 8; ++j) {
          int k2 = gk + j;
          tmp.h[j] = (gm < M && k2 < K) ? A[(size_t)gm * K + k2] : (u16)0;
        }
      }
      *(uint4*)&As[s][row * LDP + kc] = tmp.v4;
    }
  };
#endif

  auto loadTileB = [&](int s, int kb) {
    // B tile: 32 (k) x 128 (n), stored transposed Bs[n][k]
    for (int c = tid; c < 512; c += 256) {
      int k = c >> 4, nb = (c & 15) << 3;
      int gk = kb + k, gn0 = nBase + nb;
      union { u16 h[8]; uint2 v2[2]; } tmp;
      size_t eoff = (size_t)gk * N + gn0;
      if (gk < K && gn0 + 8 <= N && ((eoff & 3) == 0)) {
        const u16* p = B + eoff;
        tmp.v2[0] = *(const uint2*)p;
        tmp.v2[1] = *(const uint2*)(p + 4);
        if (gk + 32 < K)  // prime L2 for the next K-tile (global_prefetch_b8)
          __builtin_prefetch((const void*)(p + (size_t)32 * N), 0, 1);
      } else {
        for (int j = 0; j < 8; ++j) {
          int gn = gn0 + j;
          tmp.h[j] = (gk < K && gn < N) ? B[(size_t)gk * N + gn] : (u16)0;
        }
      }
#pragma unroll
      for (int j = 0; j < 8; ++j) Bs[s][(nb + j) * LDP + k] = tmp.h[j];
    }
  };

  int nk = (K + 31) >> 5;
  loadTileA(0, 0);
  loadTileB(0, 0);
#if HAVE_TDM
  if (tid < 32) __builtin_amdgcn_s_wait_tensorcnt(0);
#endif
  __syncthreads();
  for (int kt = 0; kt < nk; ++kt) {
    int cur = kt & 1;
    if (kt + 1 < nk) {
      loadTileA(cur ^ 1, (kt + 1) * 32);
      loadTileB(cur ^ 1, (kt + 1) * 32);
    }

    // A fragments: lanes 0-15 carry K {0..7,16..23}, lanes 16-31 {8..15,24..31}
    union { bf16x16 v; uint4 q[2]; } af[4], bfm[2];
#pragma unroll
    for (int mt = 0; mt < 4; ++mt) {
      const u16* p = &As[cur][(wm * 64 + mt * 16 + lr) * LDP + hi * 8];
      af[mt].q[0] = *(const uint4*)p;
      af[mt].q[1] = *(const uint4*)(p + 16);
    }
    // B fragments: lanes 0-15 carry K {0..7,16..23}, lanes 16-31 {8..15,24..31} (Bs is [n][k])
#pragma unroll
    for (int nt = 0; nt < 2; ++nt) {
      const u16* p = &Bs[cur][(wn * 32 + nt * 16 + lr) * LDP + hi * 8];
      bfm[nt].q[0] = *(const uint4*)p;
      bfm[nt].q[1] = *(const uint4*)(p + 16);
    }
#pragma unroll
    for (int mt = 0; mt < 4; ++mt)
#pragma unroll
      for (int nt = 0; nt < 2; ++nt)
        acc[mt][nt] = __builtin_amdgcn_wmma_f32_16x16x32_bf16(
            false, af[mt].v, false, bfm[nt].v, (short)0, acc[mt][nt], false, false);
#if HAVE_TDM
    if (tid < 32) __builtin_amdgcn_s_wait_tensorcnt(0);
#endif
    __syncthreads();
  }

  // ---- epilogue ----
#pragma unroll
  for (int mt = 0; mt < 4; ++mt) {
#pragma unroll
    for (int nt = 0; nt < 2; ++nt) {
      int gc = nBase + wn * 32 + nt * 16 + lr;
      if (gc >= N) continue;
      float bv = bias ? bias[gc] : 0.f;
#pragma unroll
      for (int e = 0; e < 8; ++e) {
        int gr = mBase + wm * 64 + mt * 16 + hi * 8 + e;
        if (gr >= M) continue;
        size_t idx = (size_t)gr * N + gc;
        float v = acc[mt][nt][e] + bv;
        if (mode == 0) {
          outF[idx] = v;
        } else if (mode == 1) {
          outF[idx] = v + aux[idx];
        } else if (mode == 2) {
          float g = aux[idx];
          float sg = g / (1.f + __expf(-g));
          outB[idx] = f2bf(sg * v);
        } else {
          float ge = 0.5f * v * (1.f + erff(v * 0.70710678118654752f));
          outB[idx] = f2bf(ge);
        }
      }
    }
  }
}

// ---------------------------------------------------------------------------
// Host side
// ---------------------------------------------------------------------------
static inline dim3 gemm_grid(int M, int N) { return dim3((N + 127) / 128, (M + 127) / 128, 1); }

extern "C" void kernel_launch(void* const* d_in, const int* in_sizes, int n_in,
                              void* d_out, int out_size, void* d_ws, size_t ws_size,
                              hipStream_t stream) {
  const int L = 8192, H = 1280, NH = 16, D = 4, MH = 3420;
  const int P = 3 * 2 * 14 * 14;  // 1176
  const int H4 = 4 * H;           // 5120
  const int DM = 3584;

  const float* x       = (const float*)d_in[0];
  const float* cosw    = (const float*)d_in[1];
  const float* sinw    = (const float*)d_in[2];
  const float* patch_w = (const float*)d_in[3];
  const float* norm1_w = (const float*)d_in[4];
  const float* norm2_w = (const float*)d_in[5];
  const float* qkv_w   = (const float*)d_in[6];
  const float* qkv_b   = (const float*)d_in[7];
  const float* proj_w  = (const float*)d_in[8];
  const float* proj_b  = (const float*)d_in[9];
  const float* gate_w  = (const float*)d_in[10];
  const float* gate_b  = (const float*)d_in[11];
  const float* up_w    = (const float*)d_in[12];
  const float* up_b    = (const float*)d_in[13];
  const float* down_w  = (const float*)d_in[14];
  const float* down_b  = (const float*)d_in[15];
  const float* lnq_w   = (const float*)d_in[16];
  const float* m0_w    = (const float*)d_in[17];
  const float* m0_b    = (const float*)d_in[18];
  const float* m2_w    = (const float*)d_in[19];
  const float* m2_b    = (const float*)d_in[20];
  (void)in_sizes; (void)n_in; (void)out_size; (void)ws_size;

  // ---- workspace layout ----
  char* ws = (char*)d_ws;
  size_t off = 0;
  auto take = [&](size_t bytes) {
    char* p = ws + off;
    off = (off + bytes + 255) & ~(size_t)255;
    return p;
  };
  float* A     = (float*)take((size_t)L * H * 4);      // xb / mlp out
  float* RESID = (float*)take((size_t)L * H * 4);      // residual carrier (hb / h2)
  u16*   NB    = (u16*)  take((size_t)L * H * 2);      // normed activations (bf16)
  float* QKV   = (float*)take((size_t)L * 3 * H * 4);  // qkv f32
  u16*   OB    = (u16*)  take((size_t)L * H * 2);      // attention out bf16
  float* G     = (float*)take((size_t)L * MH * 4);     // gate f32 (aliases XB16)
  u16*   ACTB  = (u16*)  take((size_t)L * MH * 2);     // swiglu / gelu bf16
  u16*   WCONV = (u16*)  take((size_t)H4 * H4 * 2);    // bf16 weight staging (max = m0_w)
  u16*   XB16  = (u16*)G;                              // alias: x bf16 only needed pre-MLP

  auto cvt = [&](const float* src, u16* dst, size_t n) {
    cvt_bf16_kernel<<<2048, 256, 0, stream>>>(src, dst, n);
  };

  // resid = 0
  fill_zero_kernel<<<2048, 256, 0, stream>>>(RESID, (size_t)L * H);

  // ---- patch embed: h = x @ patch_w ----
  cvt(x, XB16, (size_t)L * P);
  cvt(patch_w, WCONV, (size_t)P * H);
  gemm_bf16_wmma<<<gemm_grid(L, H), 256, 0, stream>>>(
      XB16, WCONV, nullptr, nullptr, A, nullptr, L, H, P, 0);

  // ---- transformer blocks ----
  for (int l = 0; l < D; ++l) {
    rmsnorm_kernel<<<L, 256, 0, stream>>>(A, RESID, norm1_w + (size_t)l * H, RESID, NB);
    cvt(qkv_w + (size_t)l * H * 3 * H, WCONV, (size_t)H * 3 * H);
    gemm_bf16_wmma<<<gemm_grid(L, 3 * H), 256, 0, stream>>>(
        NB, WCONV, qkv_b + (size_t)l * 3 * H, nullptr, QKV, nullptr, L, 3 * H, H, 0);
    attn_kernel<<<(L / 64) * NH, 64, 0, stream>>>(QKV, cosw, sinw, OB);
    cvt(proj_w + (size_t)l * H * H, WCONV, (size_t)H * H);
    gemm_bf16_wmma<<<gemm_grid(L, H), 256, 0, stream>>>(
        OB, WCONV, proj_b + (size_t)l * H, RESID, RESID, nullptr, L, H, H, 1);
    rmsnorm_kernel<<<L, 256, 0, stream>>>(RESID, nullptr, norm2_w + (size_t)l * H, nullptr, NB);
    cvt(gate_w + (size_t)l * H * MH, WCONV, (size_t)H * MH);
    gemm_bf16_wmma<<<gemm_grid(L, MH), 256, 0, stream>>>(
        NB, WCONV, gate_b + (size_t)l * MH, nullptr, G, nullptr, L, MH, H, 0);
    cvt(up_w + (size_t)l * H * MH, WCONV, (size_t)H * MH);
    gemm_bf16_wmma<<<gemm_grid(L, MH), 256, 0, stream>>>(
        NB, WCONV, up_b + (size_t)l * MH, G, nullptr, ACTB, L, MH, H, 2);
    cvt(down_w + (size_t)l * MH * H, WCONV, (size_t)MH * H);
    gemm_bf16_wmma<<<gemm_grid(L, H), 256, 0, stream>>>(
        ACTB, WCONV, down_b + (size_t)l * H, nullptr, A, nullptr, L, H, MH, 0);
  }

  // ---- patch merger ----
  rmsnorm_kernel<<<L, 256, 0, stream>>>(A, RESID, lnq_w, nullptr, NB);
  cvt(m0_w, WCONV, (size_t)H4 * H4);
  gemm_bf16_wmma<<<gemm_grid(L / 4, H4), 256, 0, stream>>>(
      NB, WCONV, m0_b, nullptr, nullptr, ACTB, L / 4, H4, H4, 3);
  cvt(m2_w, WCONV, (size_t)H4 * DM);
  gemm_bf16_wmma<<<gemm_grid(L / 4, DM), 256, 0, stream>>>(
      ACTB, WCONV, m2_b, nullptr, (float*)d_out, nullptr, L / 4, DM, H4, 0);
}
